// Attention_39651138076910
// MI455X (gfx1250) — matrix operand order repaired and checked by reading, not compiled, not run on hardware
//
#include <hip/hip_runtime.h>
#include <hip/hip_bf16.h>

typedef __attribute__((ext_vector_type(16))) _Float16 v16h;
typedef __attribute__((ext_vector_type(8)))  float    v8f;

#define B_ 4
#define H_ 16
#define S_ 2048
#define D_ 64

// ---------------- K/V pre-conversion: f32 -> f16, V transposed to [bh][d][key] ----
__global__ void __launch_bounds__(256)
convert_kv_kernel(const float* __restrict__ K, const float* __restrict__ V,
                  _Float16* __restrict__ Kh, _Float16* __restrict__ Vt) {
  const size_t i = (size_t)blockIdx.x * blockDim.x + threadIdx.x;  // < B*H*S*D
  Kh[i] = (_Float16)K[i];
  // output-indexed transpose: i -> [bh][d][key]; reads are L2 hits (V = 32MB << 192MB)
  const size_t bh  = i / ((size_t)S_ * D_);
  const size_t rem = i % ((size_t)S_ * D_);
  const size_t d   = rem / S_;
  const size_t key = rem % S_;
  Vt[i] = (_Float16)V[bh * (size_t)S_ * D_ + key * D_ + d];
}

// B layout (16-bit, 32x16): col N = lane&15, element e -> K-dim = dofs + 16*hi + e
template <bool FAST>
__device__ __forceinline__ v8f score_tile(v16h aq0, v16h aq1,
                                          const float* __restrict__ Kb,
                                          const _Float16* __restrict__ Kh,
                                          unsigned kb, unsigned c, unsigned hi) {
  const unsigned off = (kb + c) * D_ + 16u * hi;  // 32-bit: saddr + voffset form
  v16h b0, b1;
  if (FAST) {
    b0 = *(const v16h*)(Kh + off);        // 32B contiguous, aligned
    b1 = *(const v16h*)(Kh + off + 32u);
  } else {
    const float* kr = Kb + off;
#pragma unroll
    for (int e = 0; e < 16; ++e) { b0[e] = (_Float16)kr[e]; b1[e] = (_Float16)kr[e + 32]; }
  }
  v8f z = {};
  v8f s = __builtin_amdgcn_wmma_f32_16x16x32_f16(false, aq0, false, b0,
                                                 (short)0, z, false, false);
  s = __builtin_amdgcn_wmma_f32_16x16x32_f16(false, aq1, false, b1,
                                             (short)0, s, false, false);
  return s;
}

// ---------------- fused attention: one wave32 per 16-row q-tile ----------------
template <bool FAST>
__global__ void __launch_bounds__(128)
attn_fused_kernel(const float* __restrict__ Q, const float* __restrict__ K,
                  const float* __restrict__ V, const int* __restrict__ pad,
                  const _Float16* __restrict__ KhG, const _Float16* __restrict__ VtG,
                  float* __restrict__ out_res, float* __restrict__ out_wts) {
  __shared__ float sP[4][16][65];  // per-wave 16x64 P staging (C-layout -> A-layout)

  const unsigned lane = threadIdx.x & 31u;
  const unsigned w    = threadIdx.x >> 5;
  const unsigned c    = lane & 15u;
  const unsigned hi   = lane >> 4;

  const unsigned wave  = blockIdx.x * 4u + w;  // 8192 waves
  const unsigned bh    = wave >> 7;            // same-(b,h) q-tiles adjacent -> K/V in L2
  const int      qbase = (int)(wave & 127u) * 16;
  const unsigned b     = bh >> 4;

  const size_t bhSD = (size_t)bh * S_ * D_;
  const float*    Qb = Q + bhSD;
  const float*    Kb = K + bhSD;
  const float*    Vb = V + bhSD;
  const _Float16* Kh = KhG + bhSD;
  const _Float16* Vt = VtG + bhSD;  // [d][key]
  const int*      pb = pad + (size_t)b * S_;

  // Q A-fragments (16x32), 1/sqrt(D) folded. A layout: row M = c,
  // element e -> d = 16*hi + e
  v16h aq0, aq1;
  {
    const float* qr = Qb + (unsigned)(qbase + (int)c) * D_;
#pragma unroll
    for (int e = 0; e < 16; ++e) {
      int d = 16 * (int)hi + e;
      aq0[e] = (_Float16)(qr[d] * 0.125f);
      aq1[e] = (_Float16)(qr[d + 32] * 0.125f);
    }
  }

  v16h bones;
#pragma unroll
  for (int e = 0; e < 16; ++e) bones[e] = (_Float16)1.0f;

  float rm[8];
#pragma unroll
  for (int v = 0; v < 8; ++v) rm[v] = -3.0e38f;
  v8f o[4];
#pragma unroll
  for (int n = 0; n < 4; ++n) o[n] = (v8f){};
  v8f osum = {};  // row-sums of P (replicated across columns)

  // ================= Pass 1: flash attention, 64 keys / iter =================
  for (unsigned kb = 0; kb < S_; kb += 64) {
    if (FAST) {  // prefetch next tiles (global_prefetch_b8)
      const unsigned kn = (kb + 64 < S_) ? kb + 64 : kb;
      __builtin_prefetch(Kh + (kn + c) * D_ + 16u * hi, 0, 1);
      __builtin_prefetch(Vt + c * S_ + kn + 16u * hi, 0, 1);
    }
    v8f s[4];
#pragma unroll
    for (unsigned t = 0; t < 4; ++t)
      s[t] = score_tile<FAST>(aq0, aq1, Kb, Kh, kb + 16u * t, c, hi);

    const int keyc = (int)(kb + c);
    unsigned pd[4];
#pragma unroll
    for (unsigned t = 0; t < 4; ++t) pd[t] = (unsigned)(pb[keyc + 16 * (int)t] != 0);

    // masked <=> future XOR pad  (future & pad==1 is NOT masked)
    float ms[4][8];
#pragma unroll
    for (unsigned t = 0; t < 4; ++t) {
      const int key = keyc + 16 * (int)t;
#pragma unroll
      for (int v = 0; v < 8; ++v) {
        const int qrow = qbase + v + 8 * (int)hi;
        ms[t][v] = ((((key > qrow) ? 1u : 0u) ^ pd[t]) != 0u) ? -1.0e9f : s[t][v];
      }
    }

#pragma unroll
    for (int v = 0; v < 8; ++v) {
      float m = fmaxf(fmaxf(ms[0][v], ms[1][v]), fmaxf(ms[2][v], ms[3][v]));
#pragma unroll
      for (int x = 8; x >= 1; x >>= 1) m = fmaxf(m, __shfl_xor(m, x, 32));
      const float rnew = fmaxf(rm[v], m);
      const float scl  = __expf(rm[v] - rnew);
      rm[v] = rnew;
#pragma unroll
      for (unsigned t = 0; t < 4; ++t)
        sP[w][v + 8 * (int)hi][16u * t + c] = __expf(ms[t][v] - rnew);
      o[0][v] *= scl; o[1][v] *= scl; o[2][v] *= scl; o[3][v] *= scl;
      osum[v] *= scl;
    }
    asm volatile("s_wait_dscnt 0" ::: "memory");

    // P as two A-fragments (row = c, key-columns 0..31 and 32..63)
    v16h pa0, pa1;
#pragma unroll
    for (int e = 0; e < 16; ++e) {
      int kk = 16 * (int)hi + e;
      pa0[e] = (_Float16)sP[w][c][kk];
      pa1[e] = (_Float16)sP[w][c][kk + 32];
    }

    // row sums via matrix op
    osum = __builtin_amdgcn_wmma_f32_16x16x32_f16(false, pa0, false, bones,
                                                  (short)0, osum, false, false);
    osum = __builtin_amdgcn_wmma_f32_16x16x32_f16(false, pa1, false, bones,
                                                  (short)0, osum, false, false);

    // P x V: B K-dim = key, col N = 16*n + c
#pragma unroll
    for (unsigned n = 0; n < 4; ++n) {
      v16h bv0, bv1;
      if (FAST) {
        const unsigned vo = (16u * n + c) * S_ + kb + 16u * hi;
        bv0 = *(const v16h*)(Vt + vo);
        bv1 = *(const v16h*)(Vt + vo + 32u);
      } else {
#pragma unroll
        for (int e = 0; e < 16; ++e) {
          bv0[e] = (_Float16)Vb[(kb + 16u * hi + e) * D_ + 16u * n + c];
          bv1[e] = (_Float16)Vb[(kb + 32u + 16u * hi + e) * D_ + 16u * n + c];
        }
      }
      o[n] = __builtin_amdgcn_wmma_f32_16x16x32_f16(false, pa0, false, bv0,
                                                    (short)0, o[n], false, false);
      o[n] = __builtin_amdgcn_wmma_f32_16x16x32_f16(false, pa1, false, bv1,
                                                    (short)0, o[n], false, false);
    }
  }

  float inv[8];
#pragma unroll
  for (int v = 0; v < 8; ++v) inv[v] = 1.0f / osum[v];

  // write result [B,H,S,D]
#pragma unroll
  for (int v = 0; v < 8; ++v) {
    const unsigned rb = (unsigned)(qbase + v + 8 * (int)hi) * D_ + c;
    float* rp = out_res + bhSD;
    rp[rb]       = o[0][v] * inv[v];
    rp[rb + 16u] = o[1][v] * inv[v];
    rp[rb + 32u] = o[2][v] * inv[v];
    rp[rb + 48u] = o[3][v] * inv[v];
  }

  // ====== Pass 2: recompute scores, write normalized weights [B,H,S,S] ======
  float* wb = out_wts + (size_t)bh * S_ * S_;
  unsigned qrowS[8];
#pragma unroll
  for (int v = 0; v < 8; ++v) qrowS[v] = (unsigned)(qbase + v + 8 * (int)hi) * S_;

  for (unsigned kb = 0; kb < S_; kb += 32) {
    v8f s0 = score_tile<FAST>(aq0, aq1, Kb, Kh, kb, c, hi);
    v8f s1 = score_tile<FAST>(aq0, aq1, Kb, Kh, kb + 16u, c, hi);
    const int key0 = (int)(kb + c), key1 = key0 + 16;
    const unsigned pd0 = (unsigned)(pb[key0] != 0);
    const unsigned pd1 = (unsigned)(pb[key1] != 0);
#pragma unroll
    for (int v = 0; v < 8; ++v) {
      const int qrow = qbase + v + 8 * (int)hi;
      const float m0 = ((((key0 > qrow) ? 1u : 0u) ^ pd0) != 0u) ? -1.0e9f : s0[v];
      const float m1 = ((((key1 > qrow) ? 1u : 0u) ^ pd1) != 0u) ? -1.0e9f : s1[v];
      wb[qrowS[v] + (unsigned)key0] = __expf(m0 - rm[v]) * inv[v];
      wb[qrowS[v] + (unsigned)key1] = __expf(m1 - rm[v]) * inv[v];
    }
  }
}

extern "C" void kernel_launch(void* const* d_in, const int* in_sizes, int n_in,
                              void* d_out, int out_size, void* d_ws, size_t ws_size,
                              hipStream_t stream) {
  (void)in_sizes; (void)n_in; (void)out_size;
  const float* q   = (const float*)d_in[0];
  const float* k   = (const float*)d_in[1];
  const float* v   = (const float*)d_in[2];
  const int*   pad = (const int*)d_in[3];

  float* out_res = (float*)d_out;                        // [B,H,S,D]
  float* out_wts = out_res + (size_t)B_ * H_ * S_ * D_;  // [B,H,S,S]

  const size_t total = (size_t)B_ * H_ * S_ * D_;        // 8388608
  const size_t need  = total * 2 * sizeof(_Float16);     // Kh + Vt = 32 MB

  const int total_waves = B_ * H_ * (S_ / 16);  // 8192
  dim3 grid(total_waves / 4);
  dim3 block(128);

  if (d_ws && ws_size >= need) {
    _Float16* Kh = (_Float16*)d_ws;
    _Float16* Vt = Kh + total;
    convert_kv_kernel<<<(unsigned)(total / 256), 256, 0, stream>>>(k, v, Kh, Vt);
    attn_fused_kernel<true><<<grid, block, 0, stream>>>(q, k, v, pad, Kh, Vt,
                                                        out_res, out_wts);
  } else {
    attn_fused_kernel<false><<<grid, block, 0, stream>>>(q, k, v, pad, nullptr,
                                                         nullptr, out_res, out_wts);
  }
}
